// CausalSelfAttention_16295105921418
// MI455X (gfx1250) — compile-verified, matrix operand order checked by reading, not run on hardware
//
#include <hip/hip_runtime.h>

// ---------------------------------------------------------------------------
// CausalSelfAttention for MI455X (gfx1250): bf16 WMMA everywhere.
// B=2, T=2048, C=1024, H=16, hd=64.
// ---------------------------------------------------------------------------

typedef __bf16 bf16;
typedef __attribute__((ext_vector_type(16))) __bf16 v16bf;
typedef __attribute__((ext_vector_type(8)))  __bf16 v8bf;
typedef __attribute__((ext_vector_type(4)))  __bf16 v4bf;
typedef __attribute__((ext_vector_type(8)))  float  v8f;
typedef __fp16 h2 __attribute__((ext_vector_type(2)));   // matches cvt_pkrtz return

#define N_HEAD 16
#define HD     64
#define TSEQ   2048
#define BATCH  2
#define CDIM   1024

__device__ __forceinline__ bf16 f2bf(float f) {
  union { float f; unsigned u; } v; v.f = f;
  unsigned r = v.u + 0x7FFFu + ((v.u >> 16) & 1u);   // round-to-nearest-even
  union { unsigned short s; bf16 b; } o;
  o.s = (unsigned short)(r >> 16);
  return o.b;
}
__device__ __forceinline__ float bf2f(bf16 b) {
  union { bf16 b; unsigned short s; } i; i.b = b;
  union { unsigned u; float f; } v; v.u = ((unsigned)i.s) << 16;
  return v.f;
}

// Load a 16-bit WMMA A/B fragment from K-contiguous memory.
// ISA 7.12.2 (16-bit A 16x32): lane half h holds K in [8h,8h+8) and [16+8h,16+8h+8).
// B (32x16, col-per-lane) is symmetric. Two 16-byte loads per lane.
template <typename P>
__device__ __forceinline__ v16bf load_frag(const P* p, int half) {
  v8bf lo = *(const v8bf*)(p + half * 8);
  v8bf hi = *(const v8bf*)(p + 16 + half * 8);
  return __builtin_shufflevector(lo, hi, 0,1,2,3,4,5,6,7,8,9,10,11,12,13,14,15);
}

// ---------------------------------------------------------------------------
// Kernel 1: fp32 -> bf16 conversion (vectorized x4; all sizes are /4)
// ---------------------------------------------------------------------------
__global__ void cvt_f32_bf16(const float* __restrict__ in, bf16* __restrict__ out, int n) {
  int i = (blockIdx.x * blockDim.x + threadIdx.x) * 4;
  if (i < n) {
    float4 f = *(const float4*)(in + i);
    v4bf o;
    o[0] = f2bf(f.x); o[1] = f2bf(f.y); o[2] = f2bf(f.z); o[3] = f2bf(f.w);
    *(v4bf*)(out + i) = o;
  }
}

// ---------------------------------------------------------------------------
// Kernel 2/5: C[M,N](f32 accum) = A[M,K] * B[N,K]^T, bf16 operands.
// Block: 256 thr (8 waves) -> 128x128 tile. Wave: 64 rows x 32 cols,
// 8 accumulators (v8f), 8 wmma per 32-K step.
// ---------------------------------------------------------------------------
template <bool OUT_BF16>
__global__ __launch_bounds__(256)
void gemm_bf16_nt(const bf16* __restrict__ A, const bf16* __restrict__ B,
                  void* __restrict__ Cout, int M, int N, int K) {
  const int lane = threadIdx.x & 31;
  const int wave = threadIdx.x >> 5;
  const int half = lane >> 4;
  const int lm   = lane & 15;
  const int wn   = wave & 3;        // 4 waves across N
  const int wm   = wave >> 2;       // 2 waves across M
  const int m0   = blockIdx.y * 128 + wm * 64;
  const int n0   = blockIdx.x * 128 + wn * 32;

  v8f acc[4][2];
#pragma unroll
  for (int i = 0; i < 4; ++i)
#pragma unroll
    for (int j = 0; j < 2; ++j)
#pragma unroll
      for (int r = 0; r < 8; ++r) acc[i][j][r] = 0.f;

  const bf16* ap = A + (size_t)(m0 + lm) * K;
  const bf16* bp = B + (size_t)(n0 + lm) * K;

  for (int k0 = 0; k0 < K; k0 += 32) {
    __builtin_prefetch(ap + k0 + 256, 0, 1);   // global_prefetch_b8
    __builtin_prefetch(bp + k0 + 256, 0, 1);
    v16bf a[4], b[2];
#pragma unroll
    for (int i = 0; i < 4; ++i) a[i] = load_frag(ap + (size_t)i * 16 * K + k0, half);
#pragma unroll
    for (int j = 0; j < 2; ++j) b[j] = load_frag(bp + (size_t)j * 16 * K + k0, half);
#pragma unroll
    for (int i = 0; i < 4; ++i)
#pragma unroll
      for (int j = 0; j < 2; ++j)
        acc[i][j] = __builtin_amdgcn_wmma_f32_16x16x32_bf16(
            false, a[i], false, b[j], (short)0, acc[i][j], false, false);
  }

  // C/D layout: VGPR r -> row (r + 8*half), lane lm -> col
#pragma unroll
  for (int i = 0; i < 4; ++i)
#pragma unroll
    for (int j = 0; j < 2; ++j)
#pragma unroll
      for (int r = 0; r < 8; ++r) {
        int m = m0 + i * 16 + r + half * 8;
        int n = n0 + j * 16 + lm;
        if (OUT_BF16) ((bf16*)Cout)[(size_t)m * N + n] = f2bf(acc[i][j][r]);
        else          ((float*)Cout)[(size_t)m * N + n] = acc[i][j][r];
      }
}

// ---------------------------------------------------------------------------
// Kernel 3: RoPE + scatter qkv[B*T,3C] -> qh/kh [B,H,T,hd], vT [B,H,hd,T]
// rotate_half with hd=64: pair (d, d+32), angle = t * 10000^(-d/32).
// The attention scale 1/sqrt(hd)=0.125 is folded into Q here so the flash
// kernel needs no per-score multiply.
// ---------------------------------------------------------------------------
__global__ void rope_scatter(const bf16* __restrict__ qkv,
                             bf16* __restrict__ qh, bf16* __restrict__ kh,
                             bf16* __restrict__ vT) {
  int idx = blockIdx.x * blockDim.x + threadIdx.x;  // [0, B*T*H*64)
  int dd = idx & 63;
  int h  = (idx >> 6) & 15;
  int t  = (idx >> 10) & 2047;
  int b  = idx >> 21;
  const bf16* row = qkv + (size_t)(b * TSEQ + t) * 3 * CDIM;
  size_t hb = ((size_t)(b * N_HEAD + h) * TSEQ + t) * HD;

  // V transpose: vT[b,h,dd,t]
  vT[((size_t)(b * N_HEAD + h) * HD + dd) * TSEQ + t] = row[2 * CDIM + h * HD + dd];

  if (dd < 32) {
    // 10000^(-dd/32) = exp(-dd * ln(10000)/32)
    float ang = (float)t * __expf(-(float)dd * 0.28782313662425572f);
    float c, s;
    __sincosf(ang, &s, &c);
    float q0 = bf2f(row[h * HD + dd]);
    float q1 = bf2f(row[h * HD + dd + 32]);
    float k0 = bf2f(row[CDIM + h * HD + dd]);
    float k1 = bf2f(row[CDIM + h * HD + dd + 32]);
    qh[hb + dd]      = f2bf((q0 * c - q1 * s) * 0.125f);
    qh[hb + dd + 32] = f2bf((q1 * c + q0 * s) * 0.125f);
    kh[hb + dd]      = f2bf(k0 * c - k1 * s);
    kh[hb + dd + 32] = f2bf(k1 * c + k0 * s);
  }
}

// ---------------------------------------------------------------------------
// Flash attention 64-key step, templated on causal masking: the key loop is
// split into floor((q0+1)/64) fully-unmasked steps plus exactly one masked
// diagonal step, so compares/cndmasks vanish from the steady-state loop.
// ---------------------------------------------------------------------------
template <bool MASK>
__device__ __forceinline__ void flash_step(
    int k0, int q0, int half, int lm,
    const bf16* __restrict__ kb, const bf16* __restrict__ vb, bf16* pw,
    const v16bf (&aq)[2], const v16bf& ones,
    v8f (&oacc)[4], v8f& lacc, float (&mrow)[8]) {
  // ---- S = Q K^T for four 16-key tiles ----
  v8f s[4];
#pragma unroll
  for (int t = 0; t < 4; ++t) {
#pragma unroll
    for (int r = 0; r < 8; ++r) s[t][r] = 0.f;
#pragma unroll
    for (int c = 0; c < 2; ++c) {
      v16bf bk = load_frag(kb + (size_t)(k0 + t * 16 + lm) * HD + c * 32, half);
      s[t] = __builtin_amdgcn_wmma_f32_16x16x32_bf16(
          false, aq[c], false, bk, (short)0, s[t], false, false);
    }
  }

  // ---- (mask +) per-row max over the 4 tiles ----
  float mxf[8];
#pragma unroll
  for (int r = 0; r < 8; ++r) {
    int qg = q0 + r + half * 8;
    float mx = -1e30f;
#pragma unroll
    for (int t = 0; t < 4; ++t) {
      float sv = s[t][r];
      if (MASK && (k0 + t * 16 + lm > qg)) sv = -1e30f;
      s[t][r] = sv;
      mx = fmaxf(mx, sv);
    }
    mxf[r] = mx;
  }

  // ---- f16-packed butterfly max across the 16-lane half-group ----
  // (the subtracted max cancels exactly in softmax; reduced precision is safe)
  h2 pk[4];
#pragma unroll
  for (int i = 0; i < 4; ++i)
    pk[i] = __builtin_amdgcn_cvt_pkrtz(mxf[2 * i], mxf[2 * i + 1]);
#pragma unroll
  for (int off = 8; off >= 1; off >>= 1) {
#pragma unroll
    for (int i = 0; i < 4; ++i) {
      union { h2 h; int v; } u, w;
      u.h = pk[i];
      w.v = __shfl_xor(u.v, off, 32);
      pk[i] = __builtin_elementwise_max(pk[i], w.h);
    }
  }
  float mnew[8];
#pragma unroll
  for (int i = 0; i < 4; ++i) {
    mnew[2 * i]     = fmaxf(mrow[2 * i],     (float)pk[i][0]);
    mnew[2 * i + 1] = fmaxf(mrow[2 * i + 1], (float)pk[i][1]);
  }

  // ---- rescale accumulators, exponentiate, stage P to LDS ----
#pragma unroll
  for (int r = 0; r < 8; ++r) {
    float alpha = __expf(mrow[r] - mnew[r]);
    mrow[r] = mnew[r];
    lacc[r] *= alpha;
#pragma unroll
    for (int j = 0; j < 4; ++j) oacc[j][r] *= alpha;
    int m = r + half * 8;
#pragma unroll
    for (int t = 0; t < 4; ++t)
      pw[m * 64 + t * 16 + lm] = f2bf(__expf(s[t][r] - mnew[r]));
  }
  // same-wave LDS RAW: wait DS ops + compiler barrier (CDNA5 split counter)
  asm volatile("s_wait_dscnt 0" ::: "memory");

  // ---- P fragments (16x64 over keys, two K=32 chunks) ----
  v16bf pa[2];
#pragma unroll
  for (int c = 0; c < 2; ++c)
    pa[c] = load_frag(pw + lm * 64 + c * 32, half);

  // row sum l via WMMA against ones (every column = rowsum; same recurrence as O)
#pragma unroll
  for (int c = 0; c < 2; ++c)
    lacc = __builtin_amdgcn_wmma_f32_16x16x32_bf16(
        false, pa[c], false, ones, (short)0, lacc, false, false);

  // ---- O += P V against transposed V ----
#pragma unroll
  for (int j = 0; j < 4; ++j)
#pragma unroll
    for (int c = 0; c < 2; ++c) {
      v16bf bv = load_frag(vb + (size_t)(j * 16 + lm) * TSEQ + k0 + c * 32, half);
      oacc[j] = __builtin_amdgcn_wmma_f32_16x16x32_bf16(
          false, pa[c], false, bv, (short)0, oacc[j], false, false);
    }
}

// ---------------------------------------------------------------------------
// Kernel 4: flash attention. One wave per 16-row Q tile, 64 keys per step.
// ---------------------------------------------------------------------------
__global__ __launch_bounds__(128)
void flash_attn(const bf16* __restrict__ qh, const bf16* __restrict__ kh,
                const bf16* __restrict__ vT, bf16* __restrict__ ao) {
  __shared__ bf16 plds[4][16 * 64];   // per-wave 16x64 P staging (2 KB each)
  const int lane = threadIdx.x & 31;
  const int wave = threadIdx.x >> 5;
  const int half = lane >> 4;
  const int lm   = lane & 15;
  const int b = blockIdx.z, h = blockIdx.y;
  const int q0 = (blockIdx.x * 4 + wave) * 16;

  const bf16* qb = qh + (size_t)(b * N_HEAD + h) * TSEQ * HD;
  const bf16* kb = kh + (size_t)(b * N_HEAD + h) * TSEQ * HD;
  const bf16* vb = vT + (size_t)(b * N_HEAD + h) * HD * TSEQ;
  bf16* ob = ao + (size_t)b * TSEQ * CDIM + h * HD;

  // Q fragments (16 rows x 64 K), two 32-K chunks; scale pre-folded in RoPE
  v16bf aq[2];
#pragma unroll
  for (int c = 0; c < 2; ++c)
    aq[c] = load_frag(qb + (size_t)(q0 + lm) * HD + c * 32, half);

  // all-ones B fragment for the row-sum WMMA
  v16bf ones;
  {
    bf16 one = f2bf(1.0f);
#pragma unroll
    for (int e = 0; e < 16; ++e) ones[e] = one;
  }

  v8f oacc[4], lacc;
#pragma unroll
  for (int j = 0; j < 4; ++j)
#pragma unroll
    for (int r = 0; r < 8; ++r) oacc[j][r] = 0.f;
#pragma unroll
  for (int r = 0; r < 8; ++r) lacc[r] = 0.f;
  float mrow[8];
#pragma unroll
  for (int r = 0; r < 8; ++r) mrow[r] = -1e30f;

  bf16* pw = &plds[wave][0];

  // Unmasked main loop + single masked diagonal step.
  // KF = floor((q0+1)/64)*64 satisfies KF <= q0 (main steps fully causal-valid)
  // and KF+63 >= q0+15 (one tail step covers the diagonal) for q0 % 16 == 0.
  const int KF = ((q0 + 1) >> 6) << 6;
  for (int k0 = 0; k0 < KF; k0 += 64)
    flash_step<false>(k0, q0, half, lm, kb, vb, pw, aq, ones, oacc, lacc, mrow);
  flash_step<true>(KF, q0, half, lm, kb, vb, pw, aq, ones, oacc, lacc, mrow);

  // ---- normalize and store [B,T,C] bf16 ----
#pragma unroll
  for (int r = 0; r < 8; ++r) {
    float inv = 1.f / lacc[r];
    int m = q0 + r + half * 8;
#pragma unroll
    for (int j = 0; j < 4; ++j)
      ob[(size_t)m * CDIM + j * 16 + lm] = f2bf(oacc[j][r] * inv);
  }
}

// ---------------------------------------------------------------------------
// Launch
// ---------------------------------------------------------------------------
extern "C" void kernel_launch(void* const* d_in, const int* in_sizes, int n_in,
                              void* d_out, int out_size, void* d_ws, size_t ws_size,
                              hipStream_t stream) {
  const float* x      = (const float*)d_in[0];   // [B,T,C]
  const float* w_attn = (const float*)d_in[1];   // [3C,C]
  const float* w_proj = (const float*)d_in[2];   // [C,C]
  float* out = (float*)d_out;                    // [B,T,C]

  const size_t BT = (size_t)BATCH * TSEQ;        // 4096
  const size_t N_X  = BT * CDIM;                 // 4 M
  const size_t N_WA = (size_t)3 * CDIM * CDIM;   // 3 M
  const size_t N_WP = (size_t)CDIM * CDIM;       // 1 M
  const size_t N_QKV = BT * 3 * CDIM;            // 12 M

  char* w = (char*)d_ws;
  bf16* xh   = (bf16*)w;                w += N_X   * 2;
  bf16* wh   = (bf16*)w;                w += N_WA  * 2;
  bf16* ph   = (bf16*)w;                w += N_WP  * 2;
  bf16* qkvh = (bf16*)w;                w += N_QKV * 2;
  bf16* qhB  = (bf16*)w;                w += N_X * 2;   // [B,H,T,hd]
  bf16* khB  = (bf16*)w;                w += N_X * 2;   // [B,H,T,hd]
  bf16* vTB  = (bf16*)w;                w += N_X * 2;   // [B,H,hd,T]
  bf16* aoB  = (bf16*)w;                w += N_X * 2;   // [B,T,C]

  // 1) fp32 -> bf16
  cvt_f32_bf16<<<(N_X  / 4 + 255) / 256, 256, 0, stream>>>(x,      xh, (int)N_X);
  cvt_f32_bf16<<<(N_WA / 4 + 255) / 256, 256, 0, stream>>>(w_attn, wh, (int)N_WA);
  cvt_f32_bf16<<<(N_WP / 4 + 255) / 256, 256, 0, stream>>>(w_proj, ph, (int)N_WP);

  // 2) QKV = x * w_attn^T : [4096,1024] x [3072,1024]^T -> bf16 [4096,3072]
  gemm_bf16_nt<true><<<dim3(3 * CDIM / 128, BT / 128), 256, 0, stream>>>(
      xh, wh, qkvh, (int)BT, 3 * CDIM, CDIM);

  // 3) RoPE + scatter to per-head layouts (V transposed, Q pre-scaled)
  rope_scatter<<<(unsigned)(N_X / 256), 256, 0, stream>>>(qkvh, qhB, khB, vTB);

  // 4) flash attention: grid (T/64, H, B), 4 waves/block, wave = 16 q rows
  flash_attn<<<dim3(TSEQ / 64, N_HEAD, BATCH), 128, 0, stream>>>(qhB, khB, vTB, aoB);

  // 5) out = ao * w_proj^T : [4096,1024] x [1024,1024]^T -> f32 d_out
  gemm_bf16_nt<false><<<dim3(CDIM / 128, BT / 128), 256, 0, stream>>>(
      aoB, ph, out, (int)BT, CDIM, CDIM);

  (void)in_sizes; (void)n_in; (void)out_size; (void)ws_size;
}
